// VQVAETimeSeries_39135742001822
// MI455X (gfx1250) — matrix operand order repaired and checked by reading, not compiled, not run on hardware
//
#include <hip/hip_runtime.h>

typedef __attribute__((ext_vector_type(8)))  __bf16 v8bf;
typedef __attribute__((ext_vector_type(16))) __bf16 v16bf;
typedef __attribute__((ext_vector_type(8)))  float  v8f;

#define LN_EPS 1e-5f

__device__ __forceinline__ v16bf cat8(v8bf a, v8bf b) {
    return __builtin_shufflevector(a, b, 0,1,2,3,4,5,6,7,8,9,10,11,12,13,14,15);
}

// ---- CDNA5 async global->LDS copy (ASYNCcnt path), per ISA 15.18.3 ----
__device__ __forceinline__ void async_b128(const __bf16* g, __bf16* l) {
    uint32_t lds = (uint32_t)(uintptr_t)l;  // flat-LDS aperture: addr[31:0] = LDS offset
    asm volatile("global_load_async_to_lds_b128 %0, %1, off"
                 :: "v"(lds), "v"(g) : "memory");
}
__device__ __forceinline__ void wait_async0() {
    asm volatile("s_wait_asynccnt 0" ::: "memory");
}

// ---------------- conversion kernels (tiny) ----------------
__global__ void cvt_f32_bf16(const float* __restrict__ in, __bf16* __restrict__ out, int n) {
    int i = blockIdx.x * blockDim.x + threadIdx.x;
    if (i < n) out[i] = (__bf16)in[i];
}

// in: K x N (f32, row-major)  ->  out: N x K (bf16, row-major)
__global__ void transpose_f32_bf16(const float* __restrict__ in, __bf16* __restrict__ out,
                                   int K, int N) {
    int i = blockIdx.x * blockDim.x + threadIdx.x;
    if (i < K * N) {
        int n = i / K, k = i % K;
        out[i] = (__bf16)in[k * N + n];
    }
}

// 0.5*||c||^2 per code, once
__global__ void csq_kernel(const float* __restrict__ cb, float* __restrict__ csq) {
    int c = blockIdx.x * blockDim.x + threadIdx.x;
    if (c < 512) {
        float s = 0.0f;
#pragma unroll 8
        for (int k = 0; k < 64; ++k) { float v = cb[c * 64 + k]; s += v * v; }
        csq[c] = 0.5f * s;
    }
}

// ---------------- WMMA GEMM:  C = epi(A[MxK] @ Bt[NxK]^T + bias) ----------------
// 256 threads = 8 waves. Block tile: 256 (M) x 64 (N), K-step 32, double-buffered LDS
// staged with async global->LDS copies. Each wave: 32 rows x 64 cols, 8 WMMAs/k-step.
// EPI: 0 = ReLU -> bf16, 1 = bias + LayerNorm(64-token) -> bf16, 2 = bias -> f32
template<int KTOT, int NTOT, int EPI>
__global__ __launch_bounds__(256) void gemm_wmma(const __bf16* __restrict__ A,
                                                 const __bf16* __restrict__ Bt,
                                                 const float*  __restrict__ bias,
                                                 __bf16* __restrict__ outb,
                                                 float*  __restrict__ outf) {
    constexpr int NK = KTOT / 32;
    __shared__ alignas(16) __bf16 As[2][256 * 32];  // 2 x 16 KB
    __shared__ alignas(16) __bf16 Bs[2][64 * 32];   // 2 x 4 KB
    const int tid  = threadIdx.x;
    const int lane = tid & 31, wave = tid >> 5;
    const int h = lane >> 4, ml = lane & 15;
    const int m_base = blockIdx.x * 256;
    const int n_base = blockIdx.y * 64;

    // staging roles: A — thread t copies row t (64B = 4 x async b128);
    //                B — thread t copies row t>>2, 16B chunk (t&3)
    const size_t a_off = (size_t)(m_base + tid) * KTOT;
    const int b_row = tid >> 2, b_col = (tid & 3) * 8;
    const size_t b_off = (size_t)(n_base + b_row) * KTOT + b_col;

    auto stage_async = [&](int buf, int k0) {
        const __bf16* ga = A + a_off + k0;
        __bf16* la = As[buf] + tid * 32;
        async_b128(ga,      la);
        async_b128(ga + 8,  la + 8);
        async_b128(ga + 16, la + 16);
        async_b128(ga + 24, la + 24);
        async_b128(Bt + b_off + k0, Bs[buf] + b_row * 32 + b_col);
    };

    stage_async(0, 0);
    wait_async0();
    __syncthreads();

    v8f acc[2][4];
#pragma unroll
    for (int g = 0; g < 2; ++g)
#pragma unroll
        for (int j = 0; j < 4; ++j) acc[g][j] = (v8f)(0.0f);

    for (int kt = 0; kt < NK; ++kt) {
        const int cur = kt & 1;
        if (kt + 1 < NK) stage_async(cur ^ 1, (kt + 1) * 32);  // overlaps compute

        // compute: two 16-row groups per wave, shared B fragments
        const __bf16* ap0 = As[cur] + (wave * 32 + ml) * 32;
        const __bf16* ap1 = ap0 + 16 * 32;
        v16bf af0 = cat8(*(const v8bf*)(ap0 + 8 * h), *(const v8bf*)(ap0 + 16 + 8 * h));
        v16bf af1 = cat8(*(const v8bf*)(ap1 + 8 * h), *(const v8bf*)(ap1 + 16 + 8 * h));
#pragma unroll
        for (int j = 0; j < 4; ++j) {
            const __bf16* bp = Bs[cur] + (j * 16 + ml) * 32 + 8 * h;
            v16bf bfv = cat8(*(const v8bf*)bp, *(const v8bf*)(bp + 16));
            acc[0][j] = __builtin_amdgcn_wmma_f32_16x16x32_bf16(
                false, af0, false, bfv, (short)0, acc[0][j], false, false);
            acc[1][j] = __builtin_amdgcn_wmma_f32_16x16x32_bf16(
                false, af1, false, bfv, (short)0, acc[1][j], false, false);
        }
        if (kt + 1 < NK) wait_async0();  // own copies done before publishing buffer
        __syncthreads();
    }

    // -------- epilogue --------
    float bv[4];
#pragma unroll
    for (int j = 0; j < 4; ++j) bv[j] = bias[n_base + j * 16 + ml];

#pragma unroll
    for (int g = 0; g < 2; ++g) {
        const int row0 = m_base + wave * 32 + g * 16 + 8 * h;
        if (EPI == 0) {  // ReLU -> bf16
#pragma unroll
            for (int j = 0; j < 4; ++j)
#pragma unroll
                for (int r = 0; r < 8; ++r) {
                    float v = fmaxf(acc[g][j][r] + bv[j], 0.0f);
                    outb[(size_t)(row0 + r) * NTOT + n_base + j * 16 + ml] = (__bf16)v;
                }
        } else if (EPI == 1) {  // bias + LayerNorm over the 64-wide token -> bf16
            float vals[4][8];
#pragma unroll
            for (int j = 0; j < 4; ++j)
#pragma unroll
                for (int r = 0; r < 8; ++r) vals[j][r] = acc[g][j][r] + bv[j];
#pragma unroll
            for (int r = 0; r < 8; ++r) {
                float rs = vals[0][r] + vals[1][r] + vals[2][r] + vals[3][r];
                float rq = vals[0][r] * vals[0][r] + vals[1][r] * vals[1][r] +
                           vals[2][r] * vals[2][r] + vals[3][r] * vals[3][r];
#pragma unroll
                for (int m = 1; m < 16; m <<= 1) {
                    rs += __shfl_xor(rs, m, 32);
                    rq += __shfl_xor(rq, m, 32);
                }
                float mu   = rs * (1.0f / 64.0f);
                float var  = rq * (1.0f / 64.0f) - mu * mu;
                float rinv = rsqrtf(var + LN_EPS);
#pragma unroll
                for (int j = 0; j < 4; ++j)
                    outb[(size_t)(row0 + r) * NTOT + n_base + j * 16 + ml] =
                        (__bf16)((vals[j][r] - mu) * rinv);
            }
        } else {  // bias -> f32
#pragma unroll
            for (int j = 0; j < 4; ++j)
#pragma unroll
                for (int r = 0; r < 8; ++r)
                    outf[(size_t)(row0 + r) * NTOT + n_base + j * 16 + ml] =
                        acc[g][j][r] + bv[j];
        }
    }
}

// ---------------- VQ: argmin ||z-c||^2 == argmax (z.c - 0.5*|c|^2) ----------------
// 256 threads = 8 waves, 256 token rows per block (32 per wave), 512 codes, K = 64.
// Strictly-greater compare keeps the FIRST maximum -> matches argmin tie semantics.
__global__ __launch_bounds__(256) void vq_kernel(const __bf16* __restrict__ ze,
                                                 const __bf16* __restrict__ cbb,
                                                 const float*  __restrict__ cbf,
                                                 const float*  __restrict__ csq_g,
                                                 float* __restrict__ out_zq,
                                                 float* __restrict__ out_idx,
                                                 __bf16* __restrict__ zq_bf) {
    __shared__ alignas(16) __bf16 Zs[256 * 64];  // 32 KB
    __shared__ float Csq[512];                   // 2 KB
    __shared__ int   Idx[256];
    const int tid  = threadIdx.x;
    const int lane = tid & 31, wave = tid >> 5;
    const int h = lane >> 4, ml = lane & 15;
    const int row_base = blockIdx.x * 256;

    // stage z_e tile via async copies: 256x64 bf16 = 32 KB = 2048 x b128
    {
        const __bf16* src = ze + (size_t)row_base * 64;
#pragma unroll
        for (int i = tid; i < 2048; i += 256)
            async_b128(src + i * 8, Zs + i * 8);
    }
    for (int c = tid; c < 512; c += 256) Csq[c] = csq_g[c];
    wait_async0();
    __syncthreads();

    // A fragments: two 16-row groups, two K-steps each
    const __bf16* zp0 = Zs + (wave * 32 + ml) * 64;
    const __bf16* zp1 = zp0 + 16 * 64;
    v16bf af00 = cat8(*(const v8bf*)(zp0 + 8 * h),      *(const v8bf*)(zp0 + 16 + 8 * h));
    v16bf af01 = cat8(*(const v8bf*)(zp0 + 32 + 8 * h), *(const v8bf*)(zp0 + 48 + 8 * h));
    v16bf af10 = cat8(*(const v8bf*)(zp1 + 8 * h),      *(const v8bf*)(zp1 + 16 + 8 * h));
    v16bf af11 = cat8(*(const v8bf*)(zp1 + 32 + 8 * h), *(const v8bf*)(zp1 + 48 + 8 * h));

    float bval[2][8]; int bidx[2][8];
#pragma unroll
    for (int g = 0; g < 2; ++g)
#pragma unroll
        for (int r = 0; r < 8; ++r) { bval[g][r] = -3.0e38f; bidx[g][r] = 0; }

    for (int nt = 0; nt < 32; ++nt) {  // 512 codes / 16
        const __bf16* bp = cbb + (size_t)(nt * 16 + ml) * 64;
        v16bf bf0 = cat8(*(const v8bf*)(bp + 8 * h),      *(const v8bf*)(bp + 16 + 8 * h));
        v16bf bf1 = cat8(*(const v8bf*)(bp + 32 + 8 * h), *(const v8bf*)(bp + 48 + 8 * h));
        v8f acc0 = (v8f)(0.0f), acc1 = (v8f)(0.0f);
        acc0 = __builtin_amdgcn_wmma_f32_16x16x32_bf16(false, af00, false, bf0, (short)0, acc0, false, false);
        acc0 = __builtin_amdgcn_wmma_f32_16x16x32_bf16(false, af01, false, bf1, (short)0, acc0, false, false);
        acc1 = __builtin_amdgcn_wmma_f32_16x16x32_bf16(false, af10, false, bf0, (short)0, acc1, false, false);
        acc1 = __builtin_amdgcn_wmma_f32_16x16x32_bf16(false, af11, false, bf1, (short)0, acc1, false, false);
        const int n = nt * 16 + ml;
        const float sc = Csq[n];
#pragma unroll
        for (int r = 0; r < 8; ++r) {
            float v0 = acc0[r] - sc;
            if (v0 > bval[0][r]) { bval[0][r] = v0; bidx[0][r] = n; }
            float v1 = acc1[r] - sc;
            if (v1 > bval[1][r]) { bval[1][r] = v1; bidx[1][r] = n; }
        }
    }
    // cross-lane argmax within each 16-lane half (lower lane wins ties -> lower n)
#pragma unroll
    for (int g = 0; g < 2; ++g)
#pragma unroll
        for (int r = 0; r < 8; ++r) {
            float v = bval[g][r]; int ix = bidx[g][r];
#pragma unroll
            for (int m = 1; m < 16; m <<= 1) {
                float ov = __shfl_xor(v, m, 32);
                int   oi = __shfl_xor(ix, m, 32);
                if (ov > v || (ov == v && oi < ix)) { v = ov; ix = oi; }
            }
            if (ml == 0) Idx[wave * 32 + g * 16 + r + 8 * h] = ix;
        }
    __syncthreads();

    out_idx[row_base + tid] = (float)Idx[tid];
    // gather z_q rows (f32 out + bf16 for the decoder)
    for (int e = tid; e < 256 * 64; e += 256) {
        int row = e >> 6, col = e & 63;
        float v = cbf[(size_t)Idx[row] * 64 + col];
        size_t o = (size_t)(row_base + row) * 64 + col;
        out_zq[o] = v;
        zq_bf[o]  = (__bf16)v;
    }
}

// ---------------- launcher ----------------
extern "C" void kernel_launch(void* const* d_in, const int* in_sizes, int n_in,
                              void* d_out, int out_size, void* d_ws, size_t ws_size,
                              hipStream_t stream) {
    const float* x    = (const float*)d_in[0];
    const float* W1   = (const float*)d_in[1];
    const float* b1   = (const float*)d_in[2];
    const float* W2   = (const float*)d_in[3];
    const float* b2   = (const float*)d_in[4];
    const float* cb   = (const float*)d_in[5];
    const float* Wd1  = (const float*)d_in[6];
    const float* bd1  = (const float*)d_in[7];
    const float* Wd2  = (const float*)d_in[8];
    const float* bd2  = (const float*)d_in[9];

    const int B = 16384;
    char* ws = (char*)d_ws;
    size_t off = 0;
    auto take = [&](size_t bytes) { char* p = ws + off; off = (off + bytes + 255) & ~(size_t)255; return p; };
    __bf16* xbf  = (__bf16*)take((size_t)B * 512 * 2);      // 16 MB
    __bf16* w1t  = (__bf16*)take((size_t)1024 * 512 * 2);   // 1 MB  (N x K)
    __bf16* w2t  = (__bf16*)take((size_t)512 * 1024 * 2);   // 1 MB
    __bf16* wd1t = (__bf16*)take((size_t)1024 * 512 * 2);   // 1 MB
    __bf16* wd2t = (__bf16*)take((size_t)512 * 1024 * 2);   // 1 MB
    __bf16* cbb  = (__bf16*)take((size_t)512 * 64 * 2);     // 64 KB
    float*  csq  = (float*) take((size_t)512 * 4);          // 2 KB
    __bf16* hbuf = (__bf16*)take((size_t)B * 1024 * 2);     // 32 MB (reused h / hd)
    __bf16* zeb  = (__bf16*)take((size_t)B * 512 * 2);      // 16 MB
    __bf16* zqb  = (__bf16*)take((size_t)B * 512 * 2);      // 16 MB

    float* out_recon = (float*)d_out;                       // B*512
    float* out_zq    = out_recon + (size_t)B * 512;         // B*512
    float* out_idx   = out_zq    + (size_t)B * 512;         // B*8

    // conversions / precompute
    cvt_f32_bf16<<<(B * 512 + 255) / 256, 256, 0, stream>>>(x, xbf, B * 512);
    transpose_f32_bf16<<<(512 * 1024 + 255) / 256, 256, 0, stream>>>(W1,  w1t,  512, 1024);
    transpose_f32_bf16<<<(512 * 1024 + 255) / 256, 256, 0, stream>>>(W2,  w2t,  1024, 512);
    transpose_f32_bf16<<<(512 * 1024 + 255) / 256, 256, 0, stream>>>(Wd1, wd1t, 512, 1024);
    transpose_f32_bf16<<<(512 * 1024 + 255) / 256, 256, 0, stream>>>(Wd2, wd2t, 1024, 512);
    cvt_f32_bf16<<<(512 * 64 + 255) / 256, 256, 0, stream>>>(cb, cbb, 512 * 64);
    csq_kernel<<<2, 256, 0, stream>>>(cb, csq);

    // encoder
    gemm_wmma<512, 1024, 0><<<dim3(B / 256, 16), 256, 0, stream>>>(xbf, w1t, b1, hbuf, nullptr);
    gemm_wmma<1024, 512, 1><<<dim3(B / 256, 8),  256, 0, stream>>>(hbuf, w2t, b2, zeb, nullptr);
    // vector quantize (B*8 = 131072 token rows, 256 per block)
    vq_kernel<<<(B * 8) / 256, 256, 0, stream>>>(zeb, cbb, cb, csq, out_zq, out_idx, zqb);
    // decoder
    gemm_wmma<512, 1024, 0><<<dim3(B / 256, 16), 256, 0, stream>>>(zqb, wd1t, bd1, hbuf, nullptr);
    gemm_wmma<1024, 512, 2><<<dim3(B / 256, 8),  256, 0, stream>>>(hbuf, wd2t, bd2, nullptr, out_recon);
}
